// LateInteraction_63986422776277
// MI455X (gfx1250) — matrix operand order repaired and checked by reading, not compiled, not run on hardware
//
#include <hip/hip_runtime.h>
#include <hip/hip_bf16.h>

// Problem dims (fixed by reference setup_inputs)
#define I_DIM 32
#define J_DIM 32
#define S_DIM 256
#define H_DIM 768

typedef __attribute__((ext_vector_type(16))) _Float16 v16h;
typedef __attribute__((ext_vector_type(8)))  float    v8f;

// ---- workspace layout (bytes) ----
#define QBF_BYTES ((size_t)I_DIM * S_DIM * H_DIM * 2)   // 12,582,912
#define RKN_BYTES ((size_t)J_DIM * S_DIM * 4)           // 32,768
// kT right after

// ---- LDS layout for main kernel (bytes) ----
#define QS_STRIDE 776          // elems (768 + 8 pad) -> 1552B rows (16B aligned)
#define KS_STRIDE 136          // elems (128 + 8 pad) -> 272B rows (16B aligned)
#define KS_ROWS   384          // half of K staged at a time
#define D_STRIDE  257          // floats, de-aligns banks
#define QS_OFF 0
#define QS_BYTES (64 * QS_STRIDE * 2)                    //  99,328
#define KS_OFF QS_BYTES
#define KS_BYTES (KS_ROWS * KS_STRIDE * 2)               // 104,448
#define D_OFF  (KS_OFF + KS_BYTES)                       // 203,776
#define D_BYTES (64 * D_STRIDE * 4)                      //  65,792
#define SMEM_BYTES (D_OFF + D_BYTES)                     // 269,568  (< 320KB/WGP)

// ============================================================================
// prep1: L2-normalize q rows -> f16 row-major; compute k rows' inverse norms
// ============================================================================
__global__ void li_prep1(const float* __restrict__ q, const float* __restrict__ k,
                         _Float16* __restrict__ qh, float* __restrict__ rkn) {
    __shared__ float sred[256];
    const int rid = blockIdx.x;                       // 0..16383
    const int tid = threadIdx.x;
    const bool isQ = rid < I_DIM * S_DIM;
    const float* src = isQ ? (q + (size_t)rid * H_DIM)
                           : (k + (size_t)(rid - I_DIM * S_DIM) * H_DIM);
    float x0 = src[tid], x1 = src[tid + 256], x2 = src[tid + 512];
    sred[tid] = x0 * x0 + x1 * x1 + x2 * x2;
    __syncthreads();
    for (int off = 128; off > 0; off >>= 1) {
        if (tid < off) sred[tid] += sred[tid + off];
        __syncthreads();
    }
    const float rn = 1.0f / fmaxf(sqrtf(sred[0]), 1e-12f);
    if (isQ) {
        _Float16* dst = qh + (size_t)rid * H_DIM;
        dst[tid]       = (_Float16)(x0 * rn);
        dst[tid + 256] = (_Float16)(x1 * rn);
        dst[tid + 512] = (_Float16)(x2 * rn);
    } else if (tid == 0) {
        rkn[rid - I_DIM * S_DIM] = rn;
    }
}

// ============================================================================
// prep2: normalized-k transpose  kT[j][h][t] = k[j][t][h] * rkn[j][t]  (f16)
// ============================================================================
__global__ void li_prep2(const float* __restrict__ k, const float* __restrict__ rkn,
                         _Float16* __restrict__ kT) {
    __shared__ float tile[32][33];
    const int t0 = blockIdx.x * 32;
    const int h0 = blockIdx.y * 32;
    const int j  = blockIdx.z;
    const int tx = threadIdx.x, ty = threadIdx.y;
    tile[ty][tx] = k[(size_t)j * S_DIM * H_DIM + (size_t)(t0 + ty) * H_DIM + h0 + tx];
    __syncthreads();
    const float rn = rkn[j * S_DIM + t0 + tx];
    kT[(size_t)j * H_DIM * S_DIM + (size_t)(h0 + ty) * S_DIM + t0 + tx] =
        (_Float16)(tile[tx][ty] * rn);
}

// ---- fragment loaders (A: two 16B chunks +16 elems apart; B: strided gather over K rows) ----
__device__ __forceinline__ v16h ld_frag(const _Float16* p, int elems_apart) {
    union { uint4 u[2]; v16h v; } f;
    f.u[0] = *(const uint4*)(p);
    f.u[1] = *(const uint4*)(p + elems_apart);
    return f.v;
}
__device__ __forceinline__ v16h ld_fragB(const _Float16* p) {
    v16h v;
    #pragma unroll
    for (int s = 0; s < 16; ++s) v[s] = p[(s < 8 ? s : s + 8) * KS_STRIDE];
    return v;
}

// ============================================================================
// main: one WG (8 wave32) per (i, j, s-block of 64).
// Phase 1: D[64,256] = qn_blk @ knT, each wave owns a 2x2 block of 16x16 tiles
//          (1 fragment load per WMMA), K staged in two 384-row halves, software-
//          pipelined fragment prefetch.
// Phase 2: masked decayed softmax over t, scores = sum_t p*D, reduce over s.
// ============================================================================
__global__ void __launch_bounds__(256, 1)
li_main(const _Float16* __restrict__ qh, const _Float16* __restrict__ kT,
        const float* __restrict__ qmask, const float* __restrict__ kmask,
        const float* __restrict__ alpha_raw, float* __restrict__ out) {
    extern __shared__ char smem[];
    _Float16* qs = (_Float16*)(smem + QS_OFF);
    _Float16* ks = (_Float16*)(smem + KS_OFF);
    float*    Dm = (float*)(smem + D_OFF);
    // phase-2 scratch aliases the ks region (used only after final barrier)
    float* red  = (float*)(smem + KS_OFF);
    float* red2 = red + 256;
    float* kmr  = red + 512;

    const int tid  = threadIdx.x;
    const int lane = tid & 31;
    const int wave = tid >> 5;
    const int sb = blockIdx.x;        // s-block 0..3
    const int j  = blockIdx.y;
    const int i  = blockIdx.z;

    // ---- stage q block: 64 rows x 768 f16 (6144 uint4) ----
    {
        const uint4* src = (const uint4*)(qh + ((size_t)i * S_DIM + sb * 64) * H_DIM);
        #pragma unroll
        for (int c = 0; c < 24; ++c) {
            int idx = tid + c * 256;            // 0..6143
            int row = idx / 96, col = idx % 96; // 96 uint4 per row
            uint4 v = src[(size_t)row * 96 + col];
            *(uint4*)(qs + row * QS_STRIDE + col * 8) = v;
        }
    }

    // wave -> 2x2 block of 16x16 tiles
    const int s_pair = wave >> 2;              // 0..1 : rows  s_pair*32 + {0..31}
    const int t_quad = wave & 3;               // 0..3 : cols  t_quad*32 + {0..31}
    const int koff   = (lane >> 4) << 3;       // A layout: K {0..7,16..23}/{8..15,24..31}
    const _Float16* arow_p[2] = {
        qs + (s_pair * 32 +      (lane & 15)) * QS_STRIDE + koff,
        qs + (s_pair * 32 + 16 + (lane & 15)) * QS_STRIDE + koff };
    const _Float16* brow_p = ks + koff * KS_STRIDE + t_quad * 32 + (lane & 15); // B: lane = N, K {0..7,16..23}/{8..15,24..31}

    for (int tt = 0; tt < 2; ++tt) {           // t-tiles of 128 columns
        v8f acc[2][2];
        #pragma unroll
        for (int a = 0; a < 2; ++a)
            #pragma unroll
            for (int b = 0; b < 2; ++b)
                acc[a][b] = (v8f){0.f,0.f,0.f,0.f,0.f,0.f,0.f,0.f};

        for (int kh = 0; kh < 2; ++kh) {       // two 384-row K halves of kT
            __syncthreads();                    // prior compute done before ks reuse
            // ---- stage kT half-tile: 384 x 128 f16 (6144 uint4) ----
            {
                const _Float16* src = kT + (size_t)j * H_DIM * S_DIM
                                    + (size_t)kh * KS_ROWS * S_DIM + tt * 128;
                #pragma unroll
                for (int c = 0; c < 24; ++c) {
                    int idx = tid + c * 256;    // 0..6143
                    int h = idx >> 4, cg = idx & 15;  // 16 uint4 per row of 128
                    uint4 v = *(const uint4*)(src + (size_t)h * S_DIM + cg * 8);
                    *(uint4*)(ks + h * KS_STRIDE + cg * 8) = v;
                }
            }
            __syncthreads();

            const int kb = kh * 384;
            // prologue: fragments for kc = 0
            v16h af[2], bf[2];
            #pragma unroll
            for (int a = 0; a < 2; ++a) af[a] = ld_frag(arow_p[a] + kb, 16);
            #pragma unroll
            for (int b = 0; b < 2; ++b) bf[b] = ld_fragB(brow_p + b * 16);

            #pragma unroll
            for (int kc = 0; kc < 12; ++kc) {   // K = 12 * 32 per half
                v16h naf[2], nbf[2];
                if (kc < 11) {                  // prefetch next fragments
                    #pragma unroll
                    for (int a = 0; a < 2; ++a)
                        naf[a] = ld_frag(arow_p[a] + kb + (kc + 1) * 32, 16);
                    #pragma unroll
                    for (int b = 0; b < 2; ++b)
                        nbf[b] = ld_fragB(brow_p + (size_t)(kc + 1) * 32 * KS_STRIDE + b * 16);
                }
                #pragma unroll
                for (int a = 0; a < 2; ++a)
                    #pragma unroll
                    for (int b = 0; b < 2; ++b)
                        acc[a][b] = __builtin_amdgcn_wmma_f32_16x16x32_f16(
                            false, af[a], false, bf[b], (short)0, acc[a][b], false, false);
                #pragma unroll
                for (int a = 0; a < 2; ++a) af[a] = naf[a];
                #pragma unroll
                for (int b = 0; b < 2; ++b) bf[b] = nbf[b];
            }
        }

        // ---- store 2x2 f32 D tiles (C layout: VGPR r -> M=r / r+8 by lane half)
        const int r8   = (lane >> 4) << 3;
        const int colb = tt * 128 + t_quad * 32 + (lane & 15);
        #pragma unroll
        for (int a = 0; a < 2; ++a) {
            const int rowb = s_pair * 32 + a * 16 + r8;
            #pragma unroll
            for (int b = 0; b < 2; ++b)
                #pragma unroll
                for (int r = 0; r < 8; ++r)
                    Dm[(rowb + r) * D_STRIDE + colb + b * 16] = acc[a][b][r];
        }
    }
    __syncthreads();

    // =================== phase 2: decayed masked softmax ====================
    const float a0 = alpha_raw[0];
    const float alpha = fmaxf(a0, 0.0f) + log1pf(__expf(-fabsf(a0)));  // softplus

    kmr[tid] = kmask[j * S_DIM + tid];
    __syncthreads();

    const int row = tid & 63, grp = tid >> 6;   // 4 threads per s-row, 64 t each
    const int sg  = sb * 64 + row;
    const float qm_s = qmask[i * S_DIM + sg];
    const float* drow = Dm + row * D_STRIDE + grp * 64;

    float m = -3.0e38f;
    for (int c = 0; c < 64; ++c) {
        int t = grp * 64 + c;
        float d     = drow[c];
        float valid = qm_s * kmr[t];
        float dec   = d * __expf(-alpha * fabsf((float)(sg - t)));
        float msk   = dec * valid - (1.0f - valid) * 1e9f;
        m = fmaxf(m, msk);
    }
    red[tid] = m;
    __syncthreads();
    const float M = fmaxf(fmaxf(red[row], red[row + 64]),
                          fmaxf(red[row + 128], red[row + 192]));
    __syncthreads();

    float l = 0.f, w = 0.f;
    for (int c = 0; c < 64; ++c) {
        int t = grp * 64 + c;
        float d     = drow[c];
        float valid = qm_s * kmr[t];
        float dec   = d * __expf(-alpha * fabsf((float)(sg - t)));
        float msk   = dec * valid - (1.0f - valid) * 1e9f;
        float e = __expf(msk - M);
        l += e;
        w += e * d;                 // weight by RAW similarity (matches reference)
    }
    red[tid] = l;
    red2[tid] = w;
    __syncthreads();
    if (tid < 64) {
        float L = red[tid] + red[tid + 64] + red[tid + 128] + red[tid + 192];
        float W = red2[tid] + red2[tid + 64] + red2[tid + 128] + red2[tid + 192];
        red[tid] = qm_s * (W / L);  // per-row score * q_mask
    }
    __syncthreads();
    for (int off = 32; off > 0; off >>= 1) {    // sum 64 row scores
        if (tid < off) red[tid] += red[tid + off];
        __syncthreads();
    }
    // denom = clip(sum_s q_mask[i,s], 1)
    red2[tid] = qmask[i * S_DIM + tid];
    __syncthreads();
    for (int off = 128; off > 0; off >>= 1) {
        if (tid < off) red2[tid] += red2[tid + off];
        __syncthreads();
    }
    if (tid == 0)
        atomicAdd(out + i * J_DIM + j, red[0] / fmaxf(red2[0], 1.0f));
}

// ============================================================================
extern "C" void kernel_launch(void* const* d_in, const int* in_sizes, int n_in,
                              void* d_out, int out_size, void* d_ws, size_t ws_size,
                              hipStream_t stream) {
    const float* q  = (const float*)d_in[0];   // [32,256,768]
    const float* k  = (const float*)d_in[1];   // [32,256,768]
    const float* qm = (const float*)d_in[2];   // [32,256]
    const float* km = (const float*)d_in[3];   // [32,256]
    const float* ar = (const float*)d_in[4];   // scalar
    float* out = (float*)d_out;                // [32,32]

    char* ws = (char*)d_ws;
    _Float16* qh  = (_Float16*)ws;
    float*    rkn = (float*)(ws + QBF_BYTES);
    _Float16* kT  = (_Float16*)(ws + QBF_BYTES + RKN_BYTES);

    // allow >64KB dynamic LDS; deterministic, not a stream op (capture-safe)
    (void)hipFuncSetAttribute((const void*)li_main,
                              hipFuncAttributeMaxDynamicSharedMemorySize, SMEM_BYTES);

    (void)hipMemsetAsync(d_out, 0, (size_t)out_size * sizeof(float), stream);
    li_prep1<<<dim3(I_DIM * S_DIM + J_DIM * S_DIM), dim3(256), 0, stream>>>(q, k, qh, rkn);
    li_prep2<<<dim3(S_DIM / 32, H_DIM / 32, J_DIM), dim3(32, 32), 0, stream>>>(k, rkn, kT);
    li_main<<<dim3(S_DIM / 64, J_DIM, I_DIM), dim3(256), SMEM_BYTES, stream>>>(
        qh, kT, qm, km, ar, out);
}
